// CombinatorialClassifierSplit_33294586478792
// MI455X (gfx1250) — matrix operand order repaired and checked by reading, not compiled, not run on hardware
//
#include <hip/hip_runtime.h>
#include <math.h>
#include <stdint.h>

#define B_  128
#define P_  32
#define KC  100     // classes per partition (GEMM N)
#define S_  64      // split (GEMM K-reduction)
#define C_  10000
#define KP  112     // KC padded to multiple of 16
#define CCH 64      // c-chunk per block in gather kernel

typedef __attribute__((ext_vector_type(16))) _Float16 v16h;
typedef __attribute__((ext_vector_type(8)))  float    v8f;
typedef __attribute__((ext_vector_type(4)))  unsigned int u32x4;
typedef __attribute__((ext_vector_type(8)))  int          i32x8;
typedef __attribute__((ext_vector_type(4)))  int          i32x4;

// ---- fragment packing per CDNA5 ISA 7.12.2 (16-bit, wave32) ----
// A 16x32: lane L holds row M=L&15. VGPR v (elements 2v,2v+1):
//   lanes 0-15 : K = {0..7} in v0..3, {16..23} in v4..7
//   lanes 16-31: K = {8..15} in v0..3, {24..31} in v4..7
__device__ __forceinline__ v16h load_a_frag(const float* __restrict__ xrow,
                                            int lane, int ks) {
  v16h a;
  const int kb = (lane & 16) ? 8 : 0;
#pragma unroll
  for (int v = 0; v < 8; ++v) {
    const int k0 = ks + ((v < 4) ? (kb + 2 * v) : (16 + kb + 2 * (v - 4)));
    a[2 * v]     = (_Float16)xrow[k0];
    a[2 * v + 1] = (_Float16)xrow[k0 + 1];
  }
  return a;
}

// B 32x16: lane L holds column N=L&15. Same K layout as A:
//   lanes 0-15 : K = {0..7} in v0..3, {16..23} in v4..7
//   lanes 16-31: K = {8..15} in v0..3, {24..31} in v4..7
__device__ __forceinline__ v16h load_b_frag(const float* __restrict__ Wp,
                                            int n, int lane, int ks) {
  v16h bf;
  const int kb = (lane & 16) ? 8 : 0;
  const bool valid = (n < KC);
  const float* __restrict__ wrow = Wp + (size_t)(valid ? n : 0) * S_;
#pragma unroll
  for (int v = 0; v < 8; ++v) {
    const int s0 = ks + ((v < 4) ? (kb + 2 * v) : (16 + kb + 2 * (v - 4)));
    const float f0 = valid ? wrow[s0]     : 0.f;
    const float f1 = valid ? wrow[s0 + 1] : 0.f;
    bf[2 * v]     = (_Float16)f0;
    bf[2 * v + 1] = (_Float16)f1;
  }
  return bf;
}

// ---------------- Kernel 1: grouped GEMM + bias + log_softmax ----------------
// One block per partition p. 8 waves; wave w owns M-rows [16w, 16w+16).
// logits = x[:, p*64 : p*64+64] @ W[p].T + b[p]  -> log_softmax over K ->
// stored TRANSPOSED as logp_t[p][k][b] so the gather kernel reads coalesced.
__global__ __launch_bounds__(256) void
cc_gemm_logsoftmax(const float* __restrict__ x, const float* __restrict__ W,
                   const float* __restrict__ bias, float* __restrict__ logp_t) {
  __shared__ float sL[B_ * KP];   // 56 KB logits staging

  const int p    = blockIdx.x;
  const int tid  = threadIdx.x;
  const int w    = tid >> 5;
  const int lane = tid & 31;
  const int mbase = w * 16;

  // A fragments for this wave's 16 batch rows (reused across all N-tiles)
  const int brow_a = mbase + (lane & 15);
  const float* __restrict__ xrow = x + (size_t)brow_a * (P_ * S_) + (size_t)p * S_;
  const v16h a0 = load_a_frag(xrow, lane, 0);
  const v16h a1 = load_a_frag(xrow, lane, 32);

  const float* __restrict__ Wp = W + (size_t)p * KC * S_;

#pragma unroll
  for (int nt = 0; nt < KP / 16; ++nt) {
    const int n = nt * 16 + (lane & 15);
    const v16h b0 = load_b_frag(Wp, n, lane, 0);
    const v16h b1 = load_b_frag(Wp, n, lane, 32);
    v8f acc = {};
    acc = __builtin_amdgcn_wmma_f32_16x16x32_f16(false, a0, false, b0,
                                                 (short)0, acc, false, false);
    acc = __builtin_amdgcn_wmma_f32_16x16x32_f16(false, a1, false, b1,
                                                 (short)0, acc, false, false);
    const float bv = (n < KC) ? bias[p * KC + n] : 0.f;
    // C/D layout: VGPR r -> M = mbase + r (+8 for lanes 16-31), N = lane&15
#pragma unroll
    for (int r = 0; r < 8; ++r) {
      const int brow = mbase + ((lane & 16) ? 8 : 0) + r;
      sL[brow * KP + n] = acc[r] + bv;
    }
  }
  __syncthreads();

  // Row-wise log_softmax over the first KC entries; write transposed.
  if (tid < B_) {
    const float* __restrict__ row = &sL[tid * KP];
    float m = -INFINITY;
    for (int k = 0; k < KC; ++k) m = fmaxf(m, row[k]);
    float s = 0.f;
    for (int k = 0; k < KC; ++k) s += expf(row[k] - m);
    const float l = m + logf(s);
    // lanes (consecutive b) write consecutive addresses: coalesced 128B
    for (int k = 0; k < KC; ++k)
      logp_t[((size_t)p * KC + k) * B_ + tid] = row[k] - l;
  }
}

// ---------------- Kernel 2: gather-reduce over partitions ----------------
// Block handles a chunk of 64 c-values for all 128 batch rows.
// Index tile part[0:32, c0:c0+64] is fetched by the Tensor Data Mover as a
// 2D strided tile (tile 64x32, row stride C elements) directly into LDS;
// tensor_dim0=C gives hardware OOB zero-fill on the partial last block.
// Wave w: b-group = w&3 (lanes map to b), c-half = w>>2.
// Inner loop: wave-uniform index from LDS, coalesced 128B column load from L2.
__global__ __launch_bounds__(256) void
cc_gather_sum(const float* __restrict__ logp_t, const int* __restrict__ part,
              float* __restrict__ out) {
  __shared__ int   sIdx[P_ * CCH];      // 8 KB index tile (TDM destination)
  __shared__ float sOut[8][32][33];     // transpose tile, padded stride

  const int c0  = blockIdx.x * CCH;
  const int tid = threadIdx.x;
  const int w = tid >> 5, lane = tid & 31;

  if (w == 0) {
    // ---- Tensor DMA descriptor (D#), per CDNA5 ISA ----
    const unsigned long long gaddr =
        (unsigned long long)(uintptr_t)part + (unsigned long long)c0 * 4ull;
    const unsigned int ldsaddr = (unsigned int)(uintptr_t)(void*)&sIdx[0];

    u32x4 g0;
    g0[0] = 1u;                                   // count=1 (valid), user mode
    g0[1] = ldsaddr;                              // lds_addr (bytes)
    g0[2] = (unsigned int)(gaddr & 0xffffffffu);  // global_addr[31:0]
    g0[3] = (unsigned int)((gaddr >> 32) & 0x01ffffffu)  // global_addr[56:32]
          | 0x80000000u;                          // type=2 ("image")

    i32x8 g1;
    g1[0] = (int)(2u << 16);                      // data_size=4B; wg_mask=0
    g1[1] = (int)(((unsigned)C_ & 0xffffu) << 16);        // tensor_dim0[15:0]
    g1[2] = (int)((((unsigned)C_ >> 16) & 0xffffu)        // tensor_dim0[31:16]
          | ((unsigned)P_ << 16));                        // tensor_dim1[15:0]
    g1[3] = (int)((unsigned)CCH << 16);                   // tile_dim0=64
    g1[4] = (int)(unsigned)P_;                            // tile_dim1=32, tile_dim2=0
    g1[5] = (int)(unsigned)C_;                            // tensor_dim0_stride[31:0]
    g1[6] = 0;                                            // stride[47:32]=0
    g1[7] = 0;
    i32x4 gz4 = {};                                       // groups 2/3: 2D -> NULL
    i32x8 gz8 = {};                                       // trailing group (clang-23 form)

    // clang-23 / therock-10.0 form: 6 args
    __builtin_amdgcn_tensor_load_to_lds(g0, g1, gz4, gz4, gz8, 0);
    __builtin_amdgcn_s_wait_tensorcnt(0);
  }
  __syncthreads();   // publish TDM-written tile to all waves

  const int bg = w & 3, half = w >> 2;
  const int b = bg * 32 + lane;

  for (int cc = 0; cc < 32; ++cc) {
    const int cl = half * 32 + cc;
    float acc = 0.f;
#pragma unroll
    for (int p = 0; p < P_; ++p) {
      const int idx = sIdx[p * CCH + cl];                 // wave-uniform
      acc += logp_t[((size_t)p * KC + idx) * B_ + b];     // coalesced 128B
    }
    sOut[w][cc][lane] = acc;    // row = c, col = b
  }
  __syncthreads();

  // Transposed write-out: lanes now map to c -> coalesced stores
  const int c = c0 + half * 32 + lane;
  if (c < C_) {
#pragma unroll 4
    for (int bb = 0; bb < 32; ++bb)
      out[(size_t)(bg * 32 + bb) * C_ + c] = sOut[w][lane][bb];
  }
}

extern "C" void kernel_launch(void* const* d_in, const int* in_sizes, int n_in,
                              void* d_out, int out_size, void* d_ws, size_t ws_size,
                              hipStream_t stream) {
  (void)in_sizes; (void)n_in; (void)out_size; (void)ws_size;
  const float* x    = (const float*)d_in[0];   // (B, P*S) f32
  const float* W    = (const float*)d_in[1];   // (P, K, S) f32
  const float* bias = (const float*)d_in[2];   // (P, K) f32
  const int*   part = (const int*)d_in[3];     // (P, C) i32
  float* out = (float*)d_out;                  // (B, C) f32
  float* logp_t = (float*)d_ws;                // P*KC*B_ floats = 1.64 MB scratch

  cc_gemm_logsoftmax<<<dim3(P_), dim3(256), 0, stream>>>(x, W, bias, logp_t);
  cc_gather_sum<<<dim3((C_ + CCH - 1) / CCH), dim3(256), 0, stream>>>(logp_t, part, out);
}
